// BinarizedLeNet5_BN_68693706932672
// MI455X (gfx1250) — 8-bit matrix operand order repaired and checked by reading, compiled, not run on hardware
//
#include <hip/hip_runtime.h>


typedef __attribute__((ext_vector_type(8))) int v8i;

__device__ __forceinline__ v8i vzero8() {
  v8i z;
#pragma unroll
  for (int i = 0; i < 8; ++i) z[i] = 0;
  return z;
}

__device__ __forceinline__ v8i wmma_iu8(v8i a, v8i b, v8i c) {
  return __builtin_amdgcn_wmma_i32_16x16x64_iu8(true, a, true, b, c, false, false);
}

__global__ void binw_kernel(signed char* __restrict__ dst,
                            const float* __restrict__ src,
                            int N, int K, int Kpad, int total) {
  int idx = blockIdx.x * blockDim.x + threadIdx.x;
  if (idx >= total) return;
  int n = idx / Kpad;
  int k = idx - n * Kpad;
  signed char v = 0;
  if (n < N && k < K) v = (src[n * K + k] >= 0.0f) ? (signed char)1 : (signed char)-1;
  dst[idx] = v;
}

__global__ void conv1_kernel(const float* __restrict__ x,
                             const float* __restrict__ w,
                             const float* __restrict__ bias,
                             const float* __restrict__ g, const float* __restrict__ bb,
                             const float* __restrict__ mm, const float* __restrict__ vv,
                             signed char* __restrict__ s1) {
  __shared__ signed char ws[160];
  __shared__ float s_inv[6], s_beta[6], s_bias[6];
  int t = threadIdx.x;
  if (t < 150) ws[t] = (w[t] >= 0.0f) ? (signed char)1 : (signed char)-1;
  if (t < 6) {
    float iv = g[t] * rsqrtf(vv[t] + 1e-5f);
    s_inv[t] = iv;
    s_beta[t] = bb[t] - mm[t] * iv;
    s_bias[t] = bias[t];
  }
  __syncthreads();

  int gid = blockIdx.x * blockDim.x + t;
  int b = gid / 576;
  int rem = gid - b * 576;
  int oy = rem / 24, ox = rem - oy * 24;

  const float* xp = x + (size_t)b * 784 + oy * 28 + ox;
  int sx[25];
#pragma unroll
  for (int ky = 0; ky < 5; ++ky)
#pragma unroll
    for (int kx = 0; kx < 5; ++kx)
      sx[ky * 5 + kx] = (xp[ky * 28 + kx] >= 0.0f) ? 1 : -1;

#pragma unroll
  for (int c = 0; c < 6; ++c) {
    int acc = 0;
#pragma unroll
    for (int k = 0; k < 25; ++k) acc += sx[k] * (int)ws[c * 25 + k];
    float h = (float)acc + s_bias[c];
    float val = h * s_inv[c] + s_beta[c];
    s1[(size_t)(b * 6 + c) * 576 + rem] = (val >= 0.0f) ? (signed char)1 : (signed char)-1;
  }
}

__global__ __launch_bounds__(256) void conv2_kernel(
    const signed char* __restrict__ s1, const signed char* __restrict__ wb,
    const float* __restrict__ bias,
    const float* __restrict__ g, const float* __restrict__ bb,
    const float* __restrict__ mm, const float* __restrict__ vv,
    signed char* __restrict__ s2) {
  __shared__ int At4[8 * 16 * 48];
  int wid = threadIdx.x >> 5;
  int lane = threadIdx.x & 31;
  int half = lane >> 4;
  int r = lane & 15;
  signed char* At = (signed char*)(At4 + wid * 16 * 48);

  int m0 = (blockIdx.x * 8 + wid) * 16;

  {
    int mrow = m0 + r;
    int b = mrow / 400;
    int rem = mrow - b * 400;
    int oy = rem / 20, ox = rem - oy * 20;
    const signed char* base = s1 + (size_t)b * 3456;
    for (int kk = 0; kk < 96; ++kk) {
      int k = half * 96 + kk;
      signed char val = 0;
      if (k < 150) {
        int c = k / 25;
        int rr = k - c * 25;
        int ky = rr / 5, kx = rr - ky * 5;
        val = base[(c * 24 + oy + ky) * 24 + ox + kx];
      }
      At[r * 192 + k] = val;
    }
  }
  __syncthreads();

  const signed char* Bcol = wb + r * 192;
  v8i acc = vzero8();
#pragma unroll
  for (int ks = 0; ks < 3; ++ks) {
    v8i a, bf;
#pragma unroll
    for (int i = 0; i < 8; ++i) {
      int d = (i >> 1) * 4 + half * 2 + (i & 1);
      a[i] = *(const int*)(At + r * 192 + ks * 64 + d * 4);
      int kb = (i >> 1) * 16 + half * 8 + (i & 1) * 4;
      bf[i] = *(const int*)(Bcol + ks * 64 + kb);
    }
    acc = wmma_iu8(a, bf, acc);
  }

  float iv = g[r] * rsqrtf(vv[r] + 1e-5f);
  float beta = bb[r] - mm[r] * iv;
  float bs = bias[r];
#pragma unroll
  for (int j = 0; j < 8; ++j) {
    int mrow = m0 + half * 8 + j;
    int b = mrow / 400;
    int rem = mrow - b * 400;
    float h = (float)acc[j] + bs;
    float val = h * iv + beta;
    s2[(size_t)b * 6400 + r * 400 + rem] = (val >= 0.0f) ? (signed char)1 : (signed char)-1;
  }
}

template <int NT>
__global__ __launch_bounds__(256) void bgemm_kernel(
    const signed char* __restrict__ A, int lda,
    const signed char* __restrict__ Bw, int ldb,
    const float* __restrict__ bias,
    const float* __restrict__ g, const float* __restrict__ bb,
    const float* __restrict__ mm, const float* __restrict__ vv,
    int Ksteps, int Nvalid,
    signed char* __restrict__ outS, int ldo,
    float* __restrict__ outF, int ldof) {
  int wid = threadIdx.x >> 5;
  int lane = threadIdx.x & 31;
  int half = lane >> 4;
  int r = lane & 15;
  int m0 = (blockIdx.x * 8 + wid) * 16;

  const signed char* Arow = A + (size_t)(m0 + r) * lda;

  v8i acc[NT];
#pragma unroll
  for (int nt = 0; nt < NT; ++nt) acc[nt] = vzero8();

  for (int ks = 0; ks < Ksteps; ++ks) {
    int kbyte = ks * 64;
    v8i a;
#pragma unroll
    for (int i = 0; i < 8; ++i) {
      int d = (i >> 1) * 4 + half * 2 + (i & 1);
      a[i] = *(const int*)(Arow + kbyte + d * 4);
    }
#pragma unroll
    for (int nt = 0; nt < NT; ++nt) {
      const signed char* Bcol = Bw + (size_t)(nt * 16 + r) * ldb;
      v8i bf;
#pragma unroll
      for (int i = 0; i < 8; ++i) {
        int kb = (i >> 1) * 16 + half * 8 + (i & 1) * 4;
        bf[i] = *(const int*)(Bcol + kbyte + kb);
      }
      acc[nt] = wmma_iu8(a, bf, acc[nt]);
    }
  }

#pragma unroll
  for (int nt = 0; nt < NT; ++nt) {
    int col = nt * 16 + r;
    float iv = 0.0f, beta = 0.0f, bs = 0.0f;
    if (col < Nvalid) {
      iv = g[col] * rsqrtf(vv[col] + 1e-5f);
      beta = bb[col] - mm[col] * iv;
      bs = bias[col];
    }
#pragma unroll
    for (int j = 0; j < 8; ++j) {
      int row = m0 + half * 8 + j;
      float h = (float)acc[nt][j] + bs;
      float val = h * iv + beta;
      if (outS) {
        signed char sv = (col < Nvalid) ? ((val >= 0.0f) ? (signed char)1 : (signed char)-1)
                                        : (signed char)0;
        outS[(size_t)row * ldo + col] = sv;
      } else if (col < Nvalid) {
        outF[(size_t)row * ldof + col] = val;
      }
    }
  }
}

extern "C" void kernel_launch(void* const* d_in, const int* in_sizes, int n_in,
                              void* d_out, int out_size, void* d_ws, size_t ws_size,
                              hipStream_t stream) {
  const float* x       = (const float*)d_in[0];
  const float* conv1_w = (const float*)d_in[1];
  const float* conv1_b = (const float*)d_in[2];
  const float* conv2_w = (const float*)d_in[3];
  const float* conv2_b = (const float*)d_in[4];
  const float* fc1_w   = (const float*)d_in[5];
  const float* fc1_b   = (const float*)d_in[6];
  const float* fc2_w   = (const float*)d_in[7];
  const float* fc2_b   = (const float*)d_in[8];
  const float* fc3_w   = (const float*)d_in[9];
  const float* fc3_b   = (const float*)d_in[10];
  const float* bn1_g  = (const float*)d_in[11];
  const float* bn1_b  = (const float*)d_in[12];
  const float* bn1_m  = (const float*)d_in[13];
  const float* bn1_v  = (const float*)d_in[14];
  const float* bn2_g  = (const float*)d_in[15];
  const float* bn2_b  = (const float*)d_in[16];
  const float* bn2_m  = (const float*)d_in[17];
  const float* bn2_v  = (const float*)d_in[18];
  const float* bnf1_g = (const float*)d_in[19];
  const float* bnf1_b = (const float*)d_in[20];
  const float* bnf1_m = (const float*)d_in[21];
  const float* bnf1_v = (const float*)d_in[22];
  const float* bnf2_g = (const float*)d_in[23];
  const float* bnf2_b = (const float*)d_in[24];
  const float* bnf2_m = (const float*)d_in[25];
  const float* bnf2_v = (const float*)d_in[26];
  const float* bnf3_g = (const float*)d_in[27];
  const float* bnf3_b = (const float*)d_in[28];
  const float* bnf3_m = (const float*)d_in[29];
  const float* bnf3_v = (const float*)d_in[30];
  float* out = (float*)d_out;

  char* ws = (char*)d_ws;
  signed char* s1   = (signed char*)(ws + 0);
  signed char* s2   = (signed char*)(ws + 28311552ULL);
  signed char* s3   = (signed char*)(ws + 80740352ULL);
  signed char* s4   = (signed char*)(ws + 81788928ULL);
  signed char* wb1  = (signed char*)(ws + 82837504ULL);
  signed char* wb2  = (signed char*)(ws + 83656704ULL);
  signed char* wb3  = (signed char*)(ws + 83668992ULL);
  signed char* wbc2 = (signed char*)(ws + 83671040ULL);

  binw_kernel<<<(128 * 6400 + 255) / 256, 256, 0, stream>>>(wb1, fc1_w, 120, 6400, 6400, 128 * 6400);
  binw_kernel<<<(96 * 128 + 255) / 256, 256, 0, stream>>>(wb2, fc2_w, 84, 120, 128, 96 * 128);
  binw_kernel<<<(16 * 128 + 255) / 256, 256, 0, stream>>>(wb3, fc3_w, 10, 84, 128, 16 * 128);
  binw_kernel<<<(16 * 192 + 255) / 256, 256, 0, stream>>>(wbc2, conv2_w, 16, 150, 192, 16 * 192);

  conv1_kernel<<<(8192 * 576) / 256, 256, 0, stream>>>(
      x, conv1_w, conv1_b, bn1_g, bn1_b, bn1_m, bn1_v, s1);

  conv2_kernel<<<25600, 256, 0, stream>>>(
      s1, wbc2, conv2_b, bn2_g, bn2_b, bn2_m, bn2_v, s2);

  bgemm_kernel<8><<<64, 256, 0, stream>>>(
      s2, 6400, wb1, 6400, fc1_b, bnf1_g, bnf1_b, bnf1_m, bnf1_v,
      100, 120, s3, 128, nullptr, 0);

  bgemm_kernel<6><<<64, 256, 0, stream>>>(
      s3, 128, wb2, 128, fc2_b, bnf2_g, bnf2_b, bnf2_m, bnf2_v,
      2, 84, s4, 128, nullptr, 0);

  bgemm_kernel<1><<<64, 256, 0, stream>>>(
      s4, 128, wb3, 128, fc3_b, bnf3_g, bnf3_b, bnf3_m, bnf3_v,
      2, 10, nullptr, 0, out, 10);
}
